// LSTMCell_19928648254065
// MI455X (gfx1250) — compile-verified, matrix operand order checked by reading, not run on hardware
//
#include <hip/hip_runtime.h>

// ---------------------------------------------------------------------------
// LSTM (B=64, S=512, D=1024, H=1024) for gfx1250 / MI455X, round 4.
//
//  Phase 1 (parallel):  xg[s,b,:] = x[b,s,:] @ [W_in gates]   (bf16 WMMA,
//                       fp32 accumulate, 524288 waves).
//  Phase 2 (serial, 512 launches): one fused kernel per step:
//                       acc(g) = xg_t ;  acc(g) += h_t @ W_hid[g]  (WMMA)
//                       then in-register gate math -> c, h, outputs.
//      * h tile (16 rows x 1024) staged in LDS via global_load_async_to_lds
//        (guarded; sync fallback), shared by all 8 waves of the block.
//      * h double-buffered across steps (cross-block RAW hazard).
//      * prefetch at locality 3 -> WGP scope (pull into WGP$), since the
//        weights are L2-resident and only the near cache matters.
// ---------------------------------------------------------------------------

typedef __bf16 bf16;
typedef __attribute__((ext_vector_type(16))) bf16 v16bf;
typedef __attribute__((ext_vector_type(8)))  bf16 v8bf;
typedef __attribute__((ext_vector_type(4)))  bf16 v4bf;
typedef __attribute__((ext_vector_type(8)))  float v8f;
typedef __attribute__((ext_vector_type(4)))  int   v4i;

#define AS1 __attribute__((address_space(1)))
#define AS3 __attribute__((address_space(3)))

#define B_  64
#define S_  512
#define D_  1024
#define H_  1024
#define N4_ 4096              // 4 gates * H
#define LDS_STRIDE 1032       // padded row stride (elements) for LDS h tile

#if __has_builtin(__builtin_amdgcn_global_load_async_to_lds_b128) && \
    __has_builtin(__builtin_amdgcn_s_wait_asynccnt)
#define USE_ASYNC_LDS 1
#else
#define USE_ASYNC_LDS 0
#endif

// ---------------- bf16 conversion of x ----------------
__global__ void cvt_x_kernel(const float* __restrict__ x, bf16* __restrict__ xb) {
    size_t i = ((size_t)blockIdx.x * blockDim.x + threadIdx.x) * 4;
    float4 v = *(const float4*)(x + i);
    v4bf o;
    o.x = (bf16)v.x; o.y = (bf16)v.y; o.z = (bf16)v.z; o.w = (bf16)v.w;
    *(v4bf*)(xb + i) = o;
}

// ---------------- pack W[4][1024][1024] -> Wt[j=g*H+n][k] bf16 -------------
__global__ void pack_w_kernel(const float* __restrict__ W, bf16* __restrict__ Wt) {
    size_t t = (size_t)blockIdx.x * blockDim.x + threadIdx.x; // [0, 4096*1024)
    int j = (int)(t >> 10);
    int k = (int)(t & 1023);
    int g = j >> 10;
    int n = j & 1023;
    Wt[t] = (bf16)W[((size_t)g * 1024 + k) * 1024 + n];
}

// ---------------- initial state ----------------
__global__ void init_state_kernel(const float* __restrict__ h0,
                                  const float* __restrict__ c0,
                                  bf16* __restrict__ hb0,
                                  float* __restrict__ c_buf) {
    int i = blockIdx.x * blockDim.x + threadIdx.x; // [0, B_*H_)
    hb0[i]   = (bf16)h0[i];
    c_buf[i] = c0[i];
}

// ---------------- WMMA 16-bit fragment loader ----------------
// Lane l: row/col = l&15, half = l>>4; reg r holds K = (r>>2)*16 + half*8 +
// (r&3)*2 (+1) -> two contiguous 16-byte runs at offsets half*8, 16+half*8.
__device__ __forceinline__ v16bf load_frag(const bf16* __restrict__ p, int half) {
    v8bf lo = *(const v8bf*)(p + half * 8);
    v8bf hi = *(const v8bf*)(p + 16 + half * 8);
    v16bf r;
#pragma unroll
    for (int i = 0; i < 8; ++i) { r[i] = lo[i]; r[i + 8] = hi[i]; }
    return r;
}

__device__ __forceinline__ v8f wmma_bf16(v16bf a, v16bf b, v8f c) {
    return __builtin_amdgcn_wmma_f32_16x16x32_bf16(
        false, a, false, b, (short)0, c, false, false);
}

// ---------------- Phase 1: xg = x @ W_in (all steps, all gates) ------------
// 524288 tiles, one 16x16 tile per wave; K = 1024 (32 WMMAs).
__global__ __launch_bounds__(256)
void xg_gemm_kernel(const bf16* __restrict__ xb,
                    const bf16* __restrict__ Wt_in,
                    float* __restrict__ xg) {
    const int wave = threadIdx.x >> 5;
    const int lane = threadIdx.x & 31;
    const int tile = blockIdx.x * 8 + wave;   // [0, 524288)
    const int tn   = tile & 255;              // N tile over 4096
    const int sb   = tile >> 8;               // [0, 2048) = (s, tm)
    const int s    = sb >> 2;
    const int tm   = sb & 3;
    const int half = lane >> 4;
    const int llo  = lane & 15;

    const bf16* __restrict__ arow = xb + ((size_t)(tm * 16 + llo) * S_ + s) * D_;
    const bf16* __restrict__ brow = Wt_in + (size_t)(tn * 16 + llo) * D_;

    v8f acc = {};
#pragma unroll 2
    for (int kt = 0; kt < 32; ++kt) {
        // locality 3 -> WGP scope: pull into WGP$ (weights are L2-resident).
        __builtin_prefetch((const void*)(brow + kt * 32 + 256), 0, 3);
        v16bf a = load_frag(arow + kt * 32, half);
        v16bf b = load_frag(brow + kt * 32, half);
        acc = wmma_bf16(a, b, acc);
    }

    float* __restrict__ out = xg + ((size_t)s * B_ + tm * 16 + half * 8) * N4_
                                 + tn * 16 + llo;
#pragma unroll
    for (int r = 0; r < 8; ++r) out[(size_t)r * N4_] = acc[r];
}

// ---------------- Phase 2: fused recurrent GEMM + gate math ----------------
// 32 blocks x 8 waves; wave -> (tm, tn) tile, all 4 gates.  All 8 waves of a
// block share the same 16 h rows -> staged once in LDS.
__global__ __launch_bounds__(256)
void lstm_step_kernel(const bf16*  __restrict__ h_in,     // [B, H] bf16
                      const bf16*  __restrict__ Wt_hid,   // [4096][1024] bf16
                      const float* __restrict__ xg,       // [S, B, 4096] f32
                      const float* __restrict__ bias,     // [4, H] f32
                      float* __restrict__ c_buf,          // [B, H] f32
                      bf16*  __restrict__ h_out,          // [B, H] bf16
                      float* __restrict__ hs_out,         // [B, S, H] f32
                      float* __restrict__ cs_out,         // [B, S, H] f32
                      int s) {
    __shared__ bf16 hA[16 * LDS_STRIDE];                  // ~33 KB

    const int tid  = threadIdx.x;
    const int wave = tid >> 5;
    const int lane = tid & 31;
    const int tm   = blockIdx.x >> 3;          // same for all 8 waves of block
    const int tile = blockIdx.x * 8 + wave;    // [0, 256)
    const int tn   = tile & 63;                // N tile over H
    const int half = lane >> 4;
    const int llo  = lane & 15;

    // ---- stage h rows [tm*16, tm*16+16) x H into LDS (padded stride) ----
    {
        const bf16* __restrict__ src = h_in + (size_t)tm * 16 * H_;
#pragma unroll
        for (int i = 0; i < 8; ++i) {
            int c   = tid + 256 * i;           // 16-byte chunk id, [0, 2048)
            int r   = c >> 7;                  // row 0..15
            int seg = c & 127;                 // chunk within row
#if USE_ASYNC_LDS
            __builtin_amdgcn_global_load_async_to_lds_b128(
                (AS1 v4i*)(src + (size_t)r * H_ + seg * 8),
                (AS3 v4i*)(&hA[r * LDS_STRIDE + seg * 8]),
                0, 0);
#else
            *(v8bf*)(&hA[r * LDS_STRIDE + seg * 8]) =
                *(const v8bf*)(src + (size_t)r * H_ + seg * 8);
#endif
        }
#if USE_ASYNC_LDS
        __builtin_amdgcn_s_wait_asynccnt(0);
#endif
        __syncthreads();
    }

    const int colH = tn * 16 + llo;            // hidden column (B-matrix lane)
    const int rowB = tm * 16 + half * 8;       // batch row base (C/D layout)

    // ---- seed accumulators from precomputed xg ----
    v8f acc[4];
#pragma unroll
    for (int g = 0; g < 4; ++g) {
        const float* __restrict__ xgp =
            xg + ((size_t)s * B_ + rowB) * N4_ + g * H_ + colH;
#pragma unroll
        for (int r = 0; r < 8; ++r) acc[g][r] = xgp[(size_t)r * N4_];
    }

    // ---- K loop: acc[g] += h_tile @ W_hid[g]  (A frag reused 4x) ----
    const bf16* __restrict__ aLds = &hA[llo * LDS_STRIDE];
#pragma unroll 2
    for (int kt = 0; kt < 32; ++kt) {
        v16bf a = load_frag(aLds + kt * 32, half);   // ds_load_b128 x2
#pragma unroll
        for (int g = 0; g < 4; ++g) {
            const bf16* bp = Wt_hid + (size_t)(g * H_ + colH) * H_ + kt * 32;
            __builtin_prefetch((const void*)(bp + 256), 0, 3);  // WGP scope
            v16bf b = load_frag(bp, half);
            acc[g] = wmma_bf16(a, b, acc[g]);
        }
    }

    // ---- gate math, state update, outputs ----
    const float bf_ = bias[0 * H_ + colH];
    const float bi_ = bias[1 * H_ + colH];
    const float bo_ = bias[2 * H_ + colH];
    const float ba_ = bias[3 * H_ + colH];
#pragma unroll
    for (int r = 0; r < 8; ++r) {
        const int b = rowB + r;                        // batch index
        float f = 1.0f / (1.0f + __expf(-(acc[0][r] + bf_)));
        float i = 1.0f / (1.0f + __expf(-(acc[1][r] + bi_)));
        float o = 1.0f / (1.0f + __expf(-(acc[2][r] + bo_)));
        float a = tanhf(acc[3][r] + ba_);

        const size_t ci = (size_t)b * H_ + colH;
        float c_old = c_buf[ci];
        float c_new = f * c_old + i * a;
        float h_new = tanhf(c_new) * o;

        c_buf[ci] = c_new;
        h_out[ci] = (bf16)h_new;

        const size_t off = ((size_t)b * S_ + s) * H_ + colH;
        hs_out[off] = h_new;
        cs_out[off] = c_new;
    }
}

// ---------------------------------------------------------------------------
// Workspace layout (bytes), total ~593 MB:
//   xb     bf16 [B,S,D]       0         .. 67108864
//   Wt_in  bf16 [4096][1024]  67108864  .. 75497472
//   Wt_hid bf16 [4096][1024]  75497472  .. 83886080
//   hb0    bf16 [B,H]         83886080  .. 84017152
//   hb1    bf16 [B,H]         84017152  .. 84148224
//   c_buf  f32  [B,H]         84148224  .. 84410368
//   xg     f32  [S,B,4096]    84410368  .. 621281280
// ---------------------------------------------------------------------------
extern "C" void kernel_launch(void* const* d_in, const int* in_sizes, int n_in,
                              void* d_out, int out_size, void* d_ws, size_t ws_size,
                              hipStream_t stream) {
    const float* x     = (const float*)d_in[0];
    const float* h0    = (const float*)d_in[1];
    const float* c0    = (const float*)d_in[2];
    const float* W_in  = (const float*)d_in[3];
    const float* W_hid = (const float*)d_in[4];
    const float* bias  = (const float*)d_in[5];

    float* hs_out = (float*)d_out;
    float* cs_out = hs_out + (size_t)B_ * S_ * H_;

    char* ws = (char*)d_ws;
    bf16*  xb     = (bf16*)(ws);
    bf16*  Wt_in  = (bf16*)(ws + 67108864u);
    bf16*  Wt_hid = (bf16*)(ws + 75497472u);
    bf16*  hb[2]  = { (bf16*)(ws + 83886080u), (bf16*)(ws + 84017152u) };
    float* c_buf  = (float*)(ws + 84148224u);
    float* xg     = (float*)(ws + 84410368u);

    // One-time precompute (parallel phase).
    cvt_x_kernel<<<(B_ * S_ * D_) / 4 / 256, 256, 0, stream>>>(x, xb);
    pack_w_kernel<<<(N4_ * D_) / 256, 256, 0, stream>>>(W_in, Wt_in);
    pack_w_kernel<<<(N4_ * H_) / 256, 256, 0, stream>>>(W_hid, Wt_hid);
    init_state_kernel<<<(B_ * H_) / 256, 256, 0, stream>>>(h0, c0, hb[0], c_buf);
    xg_gemm_kernel<<<65536, 256, 0, stream>>>(xb, Wt_in, xg);

    // Serial scan: h double-buffered (fused kernel reads h while other blocks
    // would otherwise be overwriting it).
    for (int s = 0; s < S_; ++s) {
        lstm_step_kernel<<<32, 256, 0, stream>>>(
            hb[s & 1], Wt_hid, xg, bias, c_buf, hb[(s + 1) & 1],
            hs_out, cs_out, s);
    }
}
